// EncodedGCN_78958678769986
// MI455X (gfx1250) — matrix operand order repaired and checked by reading, not compiled, not run on hardware
//
#include <hip/hip_runtime.h>

typedef __attribute__((ext_vector_type(16))) _Float16 v16h;
typedef __attribute__((ext_vector_type(8)))  float    v8f;

// ---------------- s-value encoder: 64 -> 32 (ReLU) -> 1 ----------------
__global__ void gcn_senc_kernel(const float* __restrict__ sv,
                                const float* __restrict__ w1, const float* __restrict__ b1,
                                const float* __restrict__ w2, const float* __restrict__ b2,
                                float* __restrict__ out) {
  __shared__ float sh[32];
  int t = threadIdx.x;
  float acc = b1[t];
  #pragma unroll 8
  for (int k = 0; k < 64; ++k) acc = fmaf(sv[k], w1[k * 32 + t], acc);
  acc = acc > 0.f ? acc : 0.f;
  sh[t] = acc * w2[t];
  __syncthreads();
  if (t == 0) {
    float s = 0.f;
    #pragma unroll
    for (int i = 0; i < 32; ++i) s += sh[i];
    out[0] = s + b2[0];
  }
}

// ---------------- degree / normalization ----------------
__global__ void gcn_deg_init(float* __restrict__ deg, int n) {
  int i = blockIdx.x * blockDim.x + threadIdx.x;
  if (i < n) deg[i] = 1.0f;  // self-loop weight
}

__global__ void gcn_deg_acc(float* __restrict__ deg, const int* __restrict__ eidx,
                            const float* __restrict__ ew, int e) {
  int i = blockIdx.x * blockDim.x + threadIdx.x;
  if (i < e) atomicAdd(&deg[eidx[e + i]], ew[i]);  // col = eidx[1][i]
}

__global__ void gcn_dinv(float* __restrict__ deg, int n) {
  int i = blockIdx.x * blockDim.x + threadIdx.x;
  if (i < n) {
    float d = deg[i];
    deg[i] = d > 0.f ? __frsqrt_rn(d) : 0.f;
  }
}

// ---------------- WMMA linear 1: [x | s_enc] (K=9 pad 32) @ W1 -> 32 ----------------
__global__ void gcn_lin1_wmma(const float* __restrict__ x, const float* __restrict__ senc,
                              const float* __restrict__ W, float* __restrict__ hlin, int n) {
  int wave = threadIdx.x >> 5;
  int lane = threadIdx.x & 31;
  int tiles = (n + 15) >> 4;
  int tile = blockIdx.x * 4 + wave;
  if (tile >= tiles) return;           // wave-uniform: EXEC all-ones inside
  int m = lane & 15, g = lane >> 4;
  int node = tile * 16 + m; if (node >= n) node = n - 1;
  // A 16x32 f16: g==0 -> K=0..7 (x features), K=8 (s_enc); g==1 -> 0
  v16h a = {};
  if (g == 0) {
    #pragma unroll
    for (int i = 0; i < 8; ++i) a[i] = (_Float16)x[(size_t)node * 8 + i];
    a[8] = (_Float16)senc[0];
  }
  // B 32x16 f16: lane group g holds K = g*16 + i of column nn; K>=9 is zero
  int nn = lane & 15;
  v16h b0 = {}, b1 = {};
  if (g == 0) {
    #pragma unroll
    for (int i = 0; i < 9; ++i) {
      b0[i] = (_Float16)W[i * 32 + nn];
      b1[i] = (_Float16)W[i * 32 + 16 + nn];
    }
  }
  v8f c0 = {}, c1 = {};
  c0 = __builtin_amdgcn_wmma_f32_16x16x32_f16(false, a, false, b0, (short)0, c0, false, false);
  c1 = __builtin_amdgcn_wmma_f32_16x16x32_f16(false, a, false, b1, (short)0, c1, false, false);
  // D: lane -> col nn, VGPR v -> row v + 8*g
  if (tile * 16 + 16 <= n) {           // wave-uniform full-tile fast path
    float* __restrict__ p = hlin + (size_t)(tile * 16 + 8 * g) * 32 + nn;
    #pragma unroll
    for (int v = 0; v < 8; ++v) {
      p[(size_t)v * 32]      = c0[v];
      p[(size_t)v * 32 + 16] = c1[v];
    }
  } else {
    #pragma unroll
    for (int v = 0; v < 8; ++v) {
      int nodev = tile * 16 + v + 8 * g;
      if (nodev < n) {
        hlin[(size_t)nodev * 32 + nn]      = c0[v];
        hlin[(size_t)nodev * 32 + 16 + nn] = c1[v];
      }
    }
  }
}

// ---------------- WMMA linear 2: h1 (K=32) @ W2 -> 16 ----------------
__global__ void gcn_lin2_wmma(const float* __restrict__ h1, const float* __restrict__ W,
                              float* __restrict__ hlin, int n) {
  int wave = threadIdx.x >> 5;
  int lane = threadIdx.x & 31;
  int tiles = (n + 15) >> 4;
  int tile = blockIdx.x * 4 + wave;
  if (tile >= tiles) return;
  int m = lane & 15, g = lane >> 4;
  int node = tile * 16 + m; if (node >= n) node = n - 1;
  v16h a;
  #pragma unroll
  for (int i = 0; i < 8; ++i) {
    a[i]     = (_Float16)h1[(size_t)node * 32 + g * 8 + i];        // K = g*8+i
    a[8 + i] = (_Float16)h1[(size_t)node * 32 + 16 + g * 8 + i];   // K = 16+g*8+i
  }
  int nn = lane & 15;
  v16h b;
  #pragma unroll
  for (int i = 0; i < 16; ++i) b[i] = (_Float16)W[(g * 8 + i + (i & 8)) * 16 + nn];
  v8f c = {};
  c = __builtin_amdgcn_wmma_f32_16x16x32_f16(false, a, false, b, (short)0, c, false, false);
  if (tile * 16 + 16 <= n) {           // wave-uniform full-tile fast path
    float* __restrict__ p = hlin + (size_t)(tile * 16 + 8 * g) * 16 + nn;
    #pragma unroll
    for (int v = 0; v < 8; ++v) p[(size_t)v * 16] = c[v];
  } else {
    #pragma unroll
    for (int v = 0; v < 8; ++v) {
      int nodev = tile * 16 + v + 8 * g;
      if (nodev < n) hlin[(size_t)nodev * 16 + nn] = c[v];
    }
  }
}

// ---------------- aggregation: self-loop init, edge scatter, finalize ----------------
// float4-vectorized: logf in {5,4}; 4 consecutive features stay within one node.
__global__ void gcn_agg_init4(float4* __restrict__ agg, const float4* __restrict__ hlin,
                              const float* __restrict__ dinv, int total4, int logf) {
  int idx = blockIdx.x * blockDim.x + threadIdx.x;
  if (idx < total4) {
    int node = idx >> (logf - 2);
    float di = dinv[node];
    float s = di * di;
    float4 h = hlin[idx];
    agg[idx] = make_float4(h.x * s, h.y * s, h.z * s, h.w * s);  // self-loop msg
  }
}

__global__ void gcn_agg_edges(float* __restrict__ agg, const float* __restrict__ hlin,
                              const float* __restrict__ dinv, const int* __restrict__ eidx,
                              const float* __restrict__ ew, int e, int logf) {
  int tid = blockIdx.x * blockDim.x + threadIdx.x;
  if (tid >= (e << logf)) return;
  int ei = tid >> logf;
  int j  = tid & ((1 << logf) - 1);
  int r = eidx[ei];
  int c = eidx[e + ei];
  float nrm = dinv[r] * ew[ei] * dinv[c];
  atomicAdd(&agg[((size_t)c << logf) + j], hlin[((size_t)r << logf) + j] * nrm);
}

__global__ void gcn_final1_4(float4* __restrict__ h1, const float4* __restrict__ agg,
                             const float* __restrict__ b, int total4) {
  int idx = blockIdx.x * blockDim.x + threadIdx.x;
  if (idx < total4) {
    int jb = (idx << 2) & 31;
    float4 a = agg[idx];
    float4 r;
    float v;
    v = a.x + b[jb + 0]; r.x = v > 0.f ? v : 0.01f * v;
    v = a.y + b[jb + 1]; r.y = v > 0.f ? v : 0.01f * v;
    v = a.z + b[jb + 2]; r.z = v > 0.f ? v : 0.01f * v;
    v = a.w + b[jb + 3]; r.w = v > 0.f ? v : 0.01f * v;
    h1[idx] = r;   // leaky_relu(0.01)
  }
}

__global__ void gcn_final_out(float* __restrict__ out, const float4* __restrict__ agg,
                              const float* __restrict__ b2, const float* __restrict__ fw,
                              const float* __restrict__ fb, int n) {
  int node = blockIdx.x * blockDim.x + threadIdx.x;
  if (node < n) {
    float acc = fb[0];
    #pragma unroll
    for (int q = 0; q < 4; ++q) {
      float4 a = agg[(size_t)node * 4 + q];
      float v;
      v = a.x + b2[q * 4 + 0]; v = v > 0.f ? v : 0.01f * v; acc = fmaf(v, fw[q * 4 + 0], acc);
      v = a.y + b2[q * 4 + 1]; v = v > 0.f ? v : 0.01f * v; acc = fmaf(v, fw[q * 4 + 1], acc);
      v = a.z + b2[q * 4 + 2]; v = v > 0.f ? v : 0.01f * v; acc = fmaf(v, fw[q * 4 + 2], acc);
      v = a.w + b2[q * 4 + 3]; v = v > 0.f ? v : 0.01f * v; acc = fmaf(v, fw[q * 4 + 3], acc);
    }
    out[node] = acc;
  }
}

extern "C" void kernel_launch(void* const* d_in, const int* in_sizes, int n_in,
                              void* d_out, int out_size, void* d_ws, size_t ws_size,
                              hipStream_t stream) {
  const float* x     = (const float*)d_in[0];
  const int*   eidx  = (const int*)d_in[1];
  const float* ew    = (const float*)d_in[2];
  const float* sv    = (const float*)d_in[3];
  const float* sfc1w = (const float*)d_in[4];
  const float* sfc1b = (const float*)d_in[5];
  const float* sfc2w = (const float*)d_in[6];
  const float* sfc2b = (const float*)d_in[7];
  const float* c1w   = (const float*)d_in[8];
  const float* c1b   = (const float*)d_in[9];
  const float* c2w   = (const float*)d_in[10];
  const float* c2b   = (const float*)d_in[11];
  const float* f1w   = (const float*)d_in[12];
  const float* f1b   = (const float*)d_in[13];

  const int n = in_sizes[0] / 8;
  const int e = in_sizes[2];

  float* ws    = (float*)d_ws;
  float* senc  = ws;                         // 16 floats (pad, keeps bufA 64B-aligned)
  float* dinv  = ws + 16;                    // n
  float* bufA  = dinv + n;                   // n*32: h_lin1, then h1
  float* bufB  = bufA + (size_t)n * 32;      // n*32: agg1, then (h_lin2 | agg2)
  float* hlin2 = bufB;                       // n*16
  float* agg2  = bufB + (size_t)n * 16;      // n*16

  const int B = 256;
  int tiles = (n + 15) >> 4;
  int wblks = (tiles + 3) / 4;
  int n32_4 = (n * 32) / 4;                  // n*32 divisible by 4
  int n16_4 = (n * 16) / 4;

  gcn_senc_kernel<<<1, 32, 0, stream>>>(sv, sfc1w, sfc1b, sfc2w, sfc2b, senc);
  gcn_deg_init<<<(n + B - 1) / B, B, 0, stream>>>(dinv, n);
  gcn_deg_acc<<<(e + B - 1) / B, B, 0, stream>>>(dinv, eidx, ew, e);
  gcn_dinv<<<(n + B - 1) / B, B, 0, stream>>>(dinv, n);

  // conv1
  gcn_lin1_wmma<<<wblks, 128, 0, stream>>>(x, senc, c1w, bufA, n);
  gcn_agg_init4<<<(n32_4 + B - 1) / B, B, 0, stream>>>((float4*)bufB, (const float4*)bufA,
                                                       dinv, n32_4, 5);
  gcn_agg_edges<<<((e << 5) + B - 1) / B, B, 0, stream>>>(bufB, bufA, dinv, eidx, ew, e, 5);
  gcn_final1_4<<<(n32_4 + B - 1) / B, B, 0, stream>>>((float4*)bufA, (const float4*)bufB,
                                                      c1b, n32_4);

  // conv2
  gcn_lin2_wmma<<<wblks, 128, 0, stream>>>(bufA, c2w, hlin2, n);
  gcn_agg_init4<<<(n16_4 + B - 1) / B, B, 0, stream>>>((float4*)agg2, (const float4*)hlin2,
                                                       dinv, n16_4, 4);
  gcn_agg_edges<<<((e << 4) + B - 1) / B, B, 0, stream>>>(agg2, hlin2, dinv, eidx, ew, e, 4);

  // bias + leaky_relu + 16->1 projection, fused
  gcn_final_out<<<(n + B - 1) / B, B, 0, stream>>>((float*)d_out, (const float4*)agg2,
                                                   c2b, f1w, f1b, n);
}
